// MultiHeadHPLSTMModule1_34179349742200
// MI455X (gfx1250) — matrix operand order repaired and checked by reading, not compiled, not run on hardware
//
#include <hip/hip_runtime.h>

// ---------------------------------------------------------------------------
// MultiHeadHPLSTM for MI455X (gfx1250): bf16 WMMA GEMMs + 3-phase parallel scan
// Round 3: move-free ping-pong pipelines (in-place buffer reloads, unroll-by-2)
// ---------------------------------------------------------------------------

typedef __attribute__((ext_vector_type(16))) __bf16       v16bf;
typedef __attribute__((ext_vector_type(8)))  float        v8f;
typedef __attribute__((ext_vector_type(4)))  unsigned int u32x4;
typedef __attribute__((ext_vector_type(8)))  unsigned int u32x8;

#define S_SEQ   32768
#define NHEAD   8
#define DMODEL  512
#define DHID    64
#define TROWS   (S_SEQ * NHEAD)   /* 262144 flattened scan rows */
#define SCAN_BT 256               /* rows per scan block        */
#define NBLK    (TROWS / SCAN_BT) /* 1024                       */
#define LDO     520               /* padded LDS row (conflict-free: 4-bank row stride) */

// ---------------- helpers ----------------

static __device__ __forceinline__ unsigned short f2bf(float f) {
  unsigned int u = __builtin_bit_cast(unsigned int, f);
  unsigned int r = (u + 0x7FFFu + ((u >> 16) & 1u)) >> 16;  // RNE
  return (unsigned short)r;
}
static __device__ __forceinline__ float bf2f(unsigned short h) {
  unsigned int u = ((unsigned int)h) << 16;
  return __builtin_bit_cast(float, u);
}
static __device__ __forceinline__ float sigmoidf_(float x) {
  return 1.0f / (1.0f + __expf(-x));
}

static __device__ __forceinline__ v16bf frag_cat(u32x4 lo, u32x4 hi) {
  u32x8 t;
  t[0] = lo.x; t[1] = lo.y; t[2] = lo.z; t[3] = lo.w;
  t[4] = hi.x; t[5] = hi.y; t[6] = hi.z; t[7] = hi.w;
  return __builtin_bit_cast(v16bf, t);
}

// A fragment: 16(M) x 32(K) bf16, row-major, row stride ldk elements.
// ISA layout: lane<16 -> K = 0..7 / 16..23 ; lane>=16 -> K = 8..15 / 24..31.
static __device__ __forceinline__ v16bf load_a_frag(const unsigned short* base, int ldk) {
  const int lane = threadIdx.x & 31;
  const int row  = lane & 15;
  const int kb   = (lane >> 4) * 8;
  const unsigned short* p = base + row * ldk + kb;
  u32x4 lo = *reinterpret_cast<const u32x4*>(p);
  u32x4 hi = *reinterpret_cast<const u32x4*>(p + 16);
  return frag_cat(lo, hi);
}

// B fragment: 32(K) x 16(N), stored column-major (W^T row-major), column stride ldk.
// ISA layout: lane<16 -> K = 0..7 / 16..23 ; lane>=16 -> K = 8..15 / 24..31 (column N = lane&15).
static __device__ __forceinline__ v16bf load_b_frag(const unsigned short* base, int ldk) {
  const int lane = threadIdx.x & 31;
  const int col  = lane & 15;
  const int kb   = (lane >> 4) * 8;
  const unsigned short* p = base + col * ldk + kb;
  u32x4 lo = *reinterpret_cast<const u32x4*>(p);
  u32x4 hi = *reinterpret_cast<const u32x4*>(p + 16);
  return frag_cat(lo, hi);
}

static __device__ __forceinline__ v8f wmma_bf16(v16bf a, v16bf b, v8f c) {
  return __builtin_amdgcn_wmma_f32_16x16x32_bf16(false, a, false, b, (short)0, c,
                                                 false, false);
}

// ---------------- K0: conversions ----------------

__global__ void k_cvt_bf16(const float* __restrict__ in, unsigned short* __restrict__ out,
                           int n) {
  int i = blockIdx.x * blockDim.x + threadIdx.x;
  if (i < n) out[i] = f2bf(in[i]);
}

// out[n*K + k] = bf16(in[k*N + n])  (transpose to B-fragment layout)
__global__ void k_cvt_transpose(const float* __restrict__ in, unsigned short* __restrict__ outT,
                                int K, int N) {
  int i = blockIdx.x * blockDim.x + threadIdx.x;
  if (i >= K * N) return;
  int k = i / N, n = i % N;
  outT[n * K + k] = f2bf(in[i]);
}

// ---------------- K1: x = inputs @ W_in + b_in  (32768x512x512) ----------------
// block = 8 waves, tile M=64 N=128; wave tile 16x64 (4 C frags), K looped by 64
// (two 32-chunks, ping-pong buffered). Buffers are reloaded IN PLACE right after
// their last use, so the pipeline needs zero register-rotation moves and every
// global load is covered by the other buffer's 4 WMMAs.

__global__ __launch_bounds__(256) void k_gemm_in(const unsigned short* __restrict__ inA,
                                                 const unsigned short* __restrict__ WT,
                                                 const float* __restrict__ b_in,
                                                 unsigned short* __restrict__ xout) {
  const int lane = threadIdx.x & 31;
  const int w    = threadIdx.x >> 5;
  const int wr   = w & 3, wc = w >> 2;
  const int m0   = blockIdx.x * 64 + wr * 16;
  const int n0   = blockIdx.y * 128 + wc * 64;

  const unsigned short* aptr = inA + m0 * DMODEL;
  const unsigned short* wptr = WT + n0 * DMODEL;

  v8f acc[4];
  const v8f vz = {0.f, 0.f, 0.f, 0.f, 0.f, 0.f, 0.f, 0.f};
#pragma unroll
  for (int j = 0; j < 4; ++j) acc[j] = vz;

  // prologue: chunks 0 and 32
  v16bf a0 = load_a_frag(aptr, DMODEL);
  v16bf a1 = load_a_frag(aptr + 32, DMODEL);
  v16bf b0[4], b1[4];
#pragma unroll
  for (int j = 0; j < 4; ++j) b0[j] = load_b_frag(wptr + (16 * j) * DMODEL, DMODEL);
#pragma unroll
  for (int j = 0; j < 4; ++j) b1[j] = load_b_frag(wptr + (16 * j) * DMODEL + 32, DMODEL);

  for (int kc = 0; kc < DMODEL; kc += 64) {
    const int k2 = (kc + 64 < DMODEL) ? kc + 64 : kc;       // tail: redundant reload
    const int k3 = (kc + 96 < DMODEL) ? kc + 96 : kc + 32;
    // prefetch streamed A two chunk-pairs ahead (global_prefetch_b8, branch-free)
    __builtin_prefetch(aptr + (lane & 15) * DMODEL + kc + 128, 0, 1);

    // half A: consume buffer 0, then reload it in place with chunk k2
#pragma unroll
    for (int j = 0; j < 4; ++j) acc[j] = wmma_bf16(a0, b0[j], acc[j]);
    a0 = load_a_frag(aptr + k2, DMODEL);
#pragma unroll
    for (int j = 0; j < 4; ++j) b0[j] = load_b_frag(wptr + (16 * j) * DMODEL + k2, DMODEL);

    // half B: consume buffer 1, then reload it in place with chunk k3
#pragma unroll
    for (int j = 0; j < 4; ++j) acc[j] = wmma_bf16(a1, b1[j], acc[j]);
    a1 = load_a_frag(aptr + k3, DMODEL);
#pragma unroll
    for (int j = 0; j < 4; ++j) b1[j] = load_b_frag(wptr + (16 * j) * DMODEL + k3, DMODEL);
  }

  const int hi = lane >> 4, c0 = lane & 15;
#pragma unroll
  for (int j = 0; j < 4; ++j) {
    const int col = n0 + 16 * j + c0;
    const float bias = b_in[col];
#pragma unroll
    for (int e = 0; e < 8; ++e) {
      const int row = m0 + 8 * hi + e;
      xout[row * DMODEL + col] = f2bf(acc[j][e] + bias);
    }
  }
}

// ---------------- K2: gates ----------------
// per wave: 16 rows of x (Tx64). g = x@W_gates (12 frags), ogp = x@W_og (4 frags),
// activations in C layout, emit fg/hr (f32) + og (bf16). Fully unrolled (K=64).

__global__ __launch_bounds__(256) void k_gates(const unsigned short* __restrict__ x,
                                               const unsigned short* __restrict__ WgT,
                                               const float* __restrict__ b_g,
                                               const unsigned short* __restrict__ WoT,
                                               const float* __restrict__ b_o,
                                               float* __restrict__ fg,
                                               float* __restrict__ hr,
                                               unsigned short* __restrict__ og) {
  const int lane = threadIdx.x & 31;
  const int w    = threadIdx.x >> 5;
  const int t0   = (blockIdx.x * 8 + w) * 16;

  v8f g[12], ogp[4];
  const v8f vz = {0.f, 0.f, 0.f, 0.f, 0.f, 0.f, 0.f, 0.f};
#pragma unroll
  for (int j = 0; j < 12; ++j) g[j] = vz;
#pragma unroll
  for (int j = 0; j < 4; ++j) ogp[j] = vz;

#pragma unroll
  for (int kc = 0; kc < DHID; kc += 32) {
    v16bf a = load_a_frag(x + t0 * DHID + kc, DHID);
#pragma unroll
    for (int j = 0; j < 12; ++j)
      g[j] = wmma_bf16(a, load_b_frag(WgT + (16 * j) * DHID + kc, DHID), g[j]);
#pragma unroll
    for (int j = 0; j < 4; ++j)
      ogp[j] = wmma_bf16(a, load_b_frag(WoT + (16 * j) * DHID + kc, DHID), ogp[j]);
  }

  const int hi = lane >> 4, c0 = lane & 15;
#pragma unroll
  for (int j = 0; j < 4; ++j) {
    const int ch = 16 * j + c0;
    const float bf_ = b_g[ch];
    const float bi  = b_g[DHID + ch];
    const float bh  = b_g[2 * DHID + ch];
    const float bo  = b_o[ch];
#pragma unroll
    for (int e = 0; e < 8; ++e) {
      const int t = t0 + 8 * hi + e;
      const float fgv = sigmoidf_(g[j][e] + bf_);
      const float igv = sigmoidf_(g[4 + j][e] + bi);
      const float hv  = tanhf(g[8 + j][e] + bh);
      const float ogv = sigmoidf_(ogp[j][e] + bo);
      fg[t * DHID + ch] = fgv;
      hr[t * DHID + ch] = igv * hv;
      og[t * DHID + ch] = f2bf(ogv);
    }
  }
}

// ---------------- K3/K4/K5: parallel linear recurrence ----------------

__global__ __launch_bounds__(64) void k_scan_local(const float* __restrict__ fg,
                                                   const float* __restrict__ hr,
                                                   float* __restrict__ aggA,
                                                   float* __restrict__ aggB) {
  const int ch = threadIdx.x;
  const int b  = blockIdx.x;
  const int t0 = b * SCAN_BT;
  float A = 1.f, B = 0.f;
  for (int i = 0; i < SCAN_BT; ++i) {
    const float f = fg[(t0 + i) * DHID + ch];
    const float h = hr[(t0 + i) * DHID + ch];
    B = fmaf(B, f, h);
    A *= f;
  }
  aggA[b * DHID + ch] = A;
  aggB[b * DHID + ch] = B;
}

__global__ __launch_bounds__(64) void k_scan_blocks(const float* __restrict__ aggA,
                                                    const float* __restrict__ aggB,
                                                    float* __restrict__ cin) {
  const int ch = threadIdx.x;
  float Bp = 0.f;
  for (int b = 0; b < NBLK; ++b) {
    cin[b * DHID + ch] = Bp;
    Bp = fmaf(Bp, aggA[b * DHID + ch], aggB[b * DHID + ch]);
  }
}

__global__ __launch_bounds__(64) void k_scan_apply(const float* __restrict__ fg,
                                                   const float* __restrict__ hr,
                                                   const float* __restrict__ cin,
                                                   const unsigned short* __restrict__ og,
                                                   unsigned short* __restrict__ sbuf) {
  const int ch = threadIdx.x;
  const int b  = blockIdx.x;
  const int t0 = b * SCAN_BT;
  float c = cin[b * DHID + ch];
  for (int i = 0; i < SCAN_BT; ++i) {
    const int t = t0 + i;
    c = fmaf(c, fg[t * DHID + ch], hr[t * DHID + ch]);
    const float o = bf2f(og[t * DHID + ch]);
    sbuf[t * DHID + ch] = f2bf(o * c);
  }
}

// ---------------- K6: fused o = s@W_c + b_c ; out = o@W_out + b_out ----------------
// block owns 32 sequence rows. Stage 1: wave w = head w projects (32x64)@(64x64)
// into a 32x512 bf16 LDS tile. Stage 2: 8 waves do the 512-K final GEMM from LDS.
// Stage-2 pipeline is ordered so every buffer is reloaded in place right after its
// last use (no rotation moves): bq loads -> wmma g0 -> bp reload -> wmma g1 -> a reload.

__global__ __launch_bounds__(256) void k_out(const unsigned short* __restrict__ s,
                                             const unsigned short* __restrict__ WcT,
                                             const float* __restrict__ b_c,
                                             const unsigned short* __restrict__ WoutT,
                                             const float* __restrict__ b_out,
                                             float* __restrict__ out) {
  __shared__ unsigned short lds_o[32 * LDO];

  const int lane = threadIdx.x & 31;
  const int w    = threadIdx.x >> 5;
  const int s0   = blockIdx.x * 32;
  const int hi   = lane >> 4, c0 = lane & 15;
  const v8f vz = {0.f, 0.f, 0.f, 0.f, 0.f, 0.f, 0.f, 0.f};

  // prefetch next block's streamed s rows for this head (branch-free)
  __builtin_prefetch(s + ((s0 + 32) * NHEAD + w) * DHID + (lane & 15) * NHEAD * DHID, 0, 1);

  // ---- stage 1: per-head projection (wave w handles head h = w) ----
  {
    const int h = w;
#pragma unroll
    for (int mg = 0; mg < 2; ++mg) {
      v8f acc[4];
#pragma unroll
      for (int j = 0; j < 4; ++j) acc[j] = vz;
#pragma unroll
      for (int kc = 0; kc < DHID; kc += 32) {
        // A rows: flattened t = (s0 + mg*16 + r)*8 + h  -> row stride 8*64
        const unsigned short* abase = s + ((s0 + mg * 16) * NHEAD + h) * DHID + kc;
        v16bf a = load_a_frag(abase, NHEAD * DHID);
#pragma unroll
        for (int j = 0; j < 4; ++j)
          acc[j] = wmma_bf16(a, load_b_frag(WcT + (16 * j) * DHID + kc, DHID), acc[j]);
      }
#pragma unroll
      for (int j = 0; j < 4; ++j) {
        const float bias = b_c[16 * j + c0];
#pragma unroll
        for (int e = 0; e < 8; ++e) {
          const int sl = mg * 16 + 8 * hi + e;
          lds_o[sl * LDO + h * DHID + 16 * j + c0] = f2bf(acc[j][e] + bias);
        }
      }
    }
  }
  __syncthreads();

  // ---- stage 2: out = o @ W_out + b_out ----
  const int mg2 = w & 1;
  const int ng  = w >> 1;          // 0..3, each 128 output cols
  const int n0  = ng * 128;
  const unsigned short* lbase = &lds_o[(mg2 * 16) * LDO];
  const unsigned short* wptr  = WoutT + n0 * DMODEL;

  v8f acc2[8];
#pragma unroll
  for (int j = 0; j < 8; ++j) acc2[j] = vz;

  // bp = group-0 fragments (cols n0..n0+63) for current kc; a0 = LDS A for current kc
  v16bf a0 = load_a_frag(lbase, LDO);
  v16bf bp[4];
#pragma unroll
  for (int j = 0; j < 4; ++j) bp[j] = load_b_frag(wptr + (16 * j) * DMODEL, DMODEL);

  for (int kc = 0; kc < DMODEL; kc += 32) {
    const int kn = (kc + 32 < DMODEL) ? kc + 32 : kc;  // tail: redundant reload
    // group-1 fragments for current kc (in flight across group-0 WMMAs)
    v16bf bq[4];
#pragma unroll
    for (int j = 0; j < 4; ++j)
      bq[j] = load_b_frag(wptr + (16 * (4 + j)) * DMODEL + kc, DMODEL);
#pragma unroll
    for (int j = 0; j < 4; ++j) acc2[j] = wmma_bf16(a0, bp[j], acc2[j]);   // last use of bp
#pragma unroll
    for (int j = 0; j < 4; ++j)
      bp[j] = load_b_frag(wptr + (16 * j) * DMODEL + kn, DMODEL);          // reload in place
#pragma unroll
    for (int j = 0; j < 4; ++j) acc2[4 + j] = wmma_bf16(a0, bq[j], acc2[4 + j]); // last use of a0
    a0 = load_a_frag(lbase + kn, LDO);                                     // LDS: short latency
  }

#pragma unroll
  for (int j = 0; j < 8; ++j) {
    const int col = n0 + 16 * j + c0;
    const float bias = b_out[col];
#pragma unroll
    for (int e = 0; e < 8; ++e) {
      const int row = s0 + mg2 * 16 + 8 * hi + e;
      out[row * DMODEL + col] = acc2[j][e] + bias;
    }
  }
}

// ---------------- host launch ----------------

extern "C" void kernel_launch(void* const* d_in, const int* in_sizes, int n_in,
                              void* d_out, int out_size, void* d_ws, size_t ws_size,
                              hipStream_t stream) {
  const float* inputs  = (const float*)d_in[0];   // (32768, 512)
  const float* W_in    = (const float*)d_in[1];   // (512, 512)
  const float* b_in    = (const float*)d_in[2];   // (512,)
  const float* W_gates = (const float*)d_in[3];   // (64, 192)
  const float* b_gates = (const float*)d_in[4];   // (192,)
  const float* W_og    = (const float*)d_in[5];   // (64, 64)
  const float* b_og    = (const float*)d_in[6];   // (64,)
  const float* W_c     = (const float*)d_in[7];   // (64, 64)
  const float* b_c     = (const float*)d_in[8];   // (64,)
  const float* W_out   = (const float*)d_in[9];   // (512, 512)
  const float* b_out   = (const float*)d_in[10];  // (512,)
  float* out = (float*)d_out;

  char* ws = (char*)d_ws;
  size_t off = 0;
  auto alloc = [&](size_t bytes) {
    char* p = ws + off;
    off = (off + bytes + 255) & ~(size_t)255;
    return p;
  };

  unsigned short* in_bf   = (unsigned short*)alloc((size_t)S_SEQ * DMODEL * 2);  // also reused as sbuf
  unsigned short* x_bf    = (unsigned short*)alloc((size_t)TROWS * DHID * 2);
  float*          fg      = (float*)alloc((size_t)TROWS * DHID * 4);
  float*          hr      = (float*)alloc((size_t)TROWS * DHID * 4);
  unsigned short* og      = (unsigned short*)alloc((size_t)TROWS * DHID * 2);
  unsigned short* WinT    = (unsigned short*)alloc((size_t)DMODEL * DMODEL * 2);
  unsigned short* WgT     = (unsigned short*)alloc((size_t)DHID * 3 * DHID * 2);
  unsigned short* WogT    = (unsigned short*)alloc((size_t)DHID * DHID * 2);
  unsigned short* WcT     = (unsigned short*)alloc((size_t)DHID * DHID * 2);
  unsigned short* WoutT   = (unsigned short*)alloc((size_t)DMODEL * DMODEL * 2);
  float*          aggA    = (float*)alloc((size_t)NBLK * DHID * 4);
  float*          aggB    = (float*)alloc((size_t)NBLK * DHID * 4);
  float*          cin     = (float*)alloc((size_t)NBLK * DHID * 4);
  unsigned short* sbuf    = in_bf;  // inputs_bf dead after K1; safe alias, rewritten each call
  (void)ws_size; (void)in_sizes; (void)n_in; (void)out_size;

  // K0: precision conversion + weight transposition (to B-fragment layout)
  {
    const int n = S_SEQ * DMODEL;
    k_cvt_bf16<<<(n + 255) / 256, 256, 0, stream>>>(inputs, in_bf, n);
    k_cvt_transpose<<<(DMODEL * DMODEL + 255) / 256, 256, 0, stream>>>(W_in, WinT, DMODEL, DMODEL);
    k_cvt_transpose<<<(DHID * 3 * DHID + 255) / 256, 256, 0, stream>>>(W_gates, WgT, DHID, 3 * DHID);
    k_cvt_transpose<<<(DHID * DHID + 255) / 256, 256, 0, stream>>>(W_og, WogT, DHID, DHID);
    k_cvt_transpose<<<(DHID * DHID + 255) / 256, 256, 0, stream>>>(W_c, WcT, DHID, DHID);
    k_cvt_transpose<<<(DMODEL * DMODEL + 255) / 256, 256, 0, stream>>>(W_out, WoutT, DMODEL, DMODEL);
  }

  // K1: x = inputs @ W_in + b_in
  k_gemm_in<<<dim3(S_SEQ / 64, DMODEL / 128), 256, 0, stream>>>(in_bf, WinT, b_in, x_bf);

  // K2: gate GEMMs + activations
  k_gates<<<TROWS / (16 * 8), 256, 0, stream>>>(x_bf, WgT, b_gates, WogT, b_og, fg, hr, og);

  // K3..K5: parallel scan c_t = fg_t * c_{t-1} + hr_t ; fuse s = og * c
  k_scan_local<<<NBLK, 64, 0, stream>>>(fg, hr, aggA, aggB);
  k_scan_blocks<<<1, 64, 0, stream>>>(aggA, aggB, cin);
  k_scan_apply<<<NBLK, 64, 0, stream>>>(fg, hr, cin, og, sbuf);

  // K6: fused o = s@W_c + b_c ; out = o@W_out + b_out
  k_out<<<S_SEQ / 32, 256, 0, stream>>>(sbuf, WcT, b_c, WoutT, b_out, out);
}
